// LoE_11982958756443
// MI455X (gfx1250) — matrix operand order repaired and checked by reading, not compiled, not run on hardware
//
#include <hip/hip_runtime.h>

typedef __attribute__((ext_vector_type(16))) _Float16 v16h;
typedef __attribute__((ext_vector_type(8)))  _Float16 v8h;
typedef __attribute__((ext_vector_type(8)))  float    v8f;
typedef __attribute__((ext_vector_type(4)))  unsigned int v4u;
typedef __attribute__((ext_vector_type(8)))  int      v8i;
typedef __attribute__((ext_vector_type(4)))  int      v4i;

#define MB   128          // pixels (rows) per workgroup
#define HDIM 256          // hidden width
#define AP   264          // LDS activation row stride (f16), 16B-aligned, conflict-padded
#define WP   72           // LDS weight-chunk row stride (f16): 64 data + 8 pad (TDM pad unit)

union Frag { v16h v; v8h h[2]; };

// ---------------------------------------------------------------------------
// Weight pre-conversion: w[t][K][256] f32  ->  Wt[t][256][Kp] f16 (transposed)
// grid = (Kp/32, 8, T), block = (32, 8). Coalesced on both sides via LDS tile.
// ---------------------------------------------------------------------------
__global__ __launch_bounds__(256) void convert_transpose_kernel(
    const float* __restrict__ w, _Float16* __restrict__ out, int K, int Kp)
{
    __shared__ float t[32][33];
    const int tx = threadIdx.x, ty = threadIdx.y;
    const int k0 = blockIdx.x * 32, n0 = blockIdx.y * 32;
    const float*  src = w   + (size_t)blockIdx.z * K  * HDIM;
    _Float16*     dst = out + (size_t)blockIdx.z * Kp * HDIM;
#pragma unroll
    for (int j = 0; j < 4; ++j) {
        const int k = k0 + ty * 4 + j;
        t[ty * 4 + j][tx] = (k < K) ? src[(size_t)k * HDIM + n0 + tx] : 0.0f;
    }
    __syncthreads();
#pragma unroll
    for (int j = 0; j < 4; ++j) {
        const int n = n0 + ty * 4 + j;
        dst[(size_t)n * Kp + k0 + tx] = (_Float16)t[tx][ty * 4 + j];
    }
}

// ---------------------------------------------------------------------------
// TDM: 2D tensor_load_to_lds of one weight K-chunk:
//   tile = 256 rows x 32 dwords (64 f16), row stride Kdw dwords,
//   LDS padding: +4 dwords every 32 dwords  ->  LDS row stride 72 f16 (== WP).
// Descriptor layout per CDNA5 ISA; 6-arg builtin.
// ---------------------------------------------------------------------------
__device__ __forceinline__ void tdm_issue_chunk(const _Float16* gsrc,
                                                unsigned lds_off, int Kdw)
{
    const unsigned long long ga = (unsigned long long)gsrc;
    v4u g0 = {};
    g0.x = 1u;                                                    // count=1, user mode
    g0.y = lds_off;                                               // lds_addr (bytes)
    g0.z = (unsigned)ga;                                          // global_addr[31:0]
    g0.w = (unsigned)((ga >> 32) & 0x1FFFFFFull) | (2u << 30);    // addr[56:32] | type=2
    v8i g1 = {};
    // data_size=4B (2), pad_enable, pad_interval=32 dw (code 4), pad_amount=4 dw (code 3)
    g1[0] = (int)((2u << 16) | (1u << 20) | (4u << 22) | (3u << 25));
    g1[1] = (int)(((unsigned)Kdw & 0xFFFFu) << 16);               // tensor_dim0[15:0]
    g1[2] = (int)((((unsigned)Kdw >> 16) & 0xFFFFu) | (256u << 16)); // dim0 hi | dim1 lo
    g1[3] = (int)(32u << 16);                                     // dim1 hi=0 | tile_dim0=32
    g1[4] = (int)256;                                             // tile_dim1=256, tile_dim2=0
    g1[5] = Kdw;                                                  // tensor_dim0_stride[31:0]
    g1[6] = 0;
    g1[7] = 0;
    v4i g2 = {};
    v4i g3 = {};
    v8i g4 = {};
    __builtin_amdgcn_tensor_load_to_lds(g0, g1, g2, g3, g4, 0);
}

// ---------------------------------------------------------------------------
// Inner GEMM step shared by both paths. Wave layout over the 128x256 block:
//   4 M-groups (32 rows = 2 m-tiles) x 2 N-groups (128 cols = 8 n-tiles).
// Each B fragment feeds two back-to-back WMMAs (same B, different A/acc).
// ---------------------------------------------------------------------------
__device__ __forceinline__ void gemm_ksteps(
    const _Float16* sA, const _Float16* sw, v8f acc[2][8],
    int cBase, int m0, int nBase, int l15, int half)
{
#pragma unroll
    for (int ks = 0; ks < 2; ++ks) {
        const int kA = cBase + ks * 32;
        Frag a0, a1;
        const _Float16* pa0 = sA + (m0 + l15) * AP + kA + 8 * half;
        a0.h[0] = *(const v8h*)(pa0);
        a0.h[1] = *(const v8h*)(pa0 + 16);
        const _Float16* pa1 = pa0 + 16 * AP;
        a1.h[0] = *(const v8h*)(pa1);
        a1.h[1] = *(const v8h*)(pa1 + 16);
#pragma unroll
        for (int nt = 0; nt < 8; ++nt) {
            Frag bf;
            const _Float16* pb = sw + (nBase + nt * 16 + l15) * WP + ks * 32 + 8 * half;
            bf.h[0] = *(const v8h*)(pb);
            bf.h[1] = *(const v8h*)(pb + 16);
            acc[0][nt] = __builtin_amdgcn_wmma_f32_16x16x32_f16(
                false, a0.v, false, bf.v, (short)0, acc[0][nt], false, false);
            acc[1][nt] = __builtin_amdgcn_wmma_f32_16x16x32_f16(
                false, a1.v, false, bf.v, (short)0, acc[1][nt], false, false);
        }
    }
}

__device__ __forceinline__ void writeback(
    _Float16* sA, const float* sBias, v8f acc[2][8],
    int m0, int nBase, int l15, int half)
{
#pragma unroll
    for (int mi = 0; mi < 2; ++mi) {
#pragma unroll
        for (int nt = 0; nt < 8; ++nt) {
            const int n = nBase + nt * 16 + l15;
            const float bias = sBias[n];
#pragma unroll
            for (int r = 0; r < 8; ++r) {
                float v = acc[mi][nt][r] + bias;
                v = v > 0.0f ? v : 0.2f * v;
                sA[(m0 + mi * 16 + r + 8 * half) * AP + n] = (_Float16)v;
            }
        }
    }
}

// ---------------------------------------------------------------------------
// One hidden layer, TDM-staged weights (pre-converted f16 Wt[256][Kp] tiles).
// Double-buffered LDS weight chunks; DMA of chunk c+1 overlaps WMMA of chunk c.
// ---------------------------------------------------------------------------
__device__ __forceinline__ void run_layer_tdm(
    _Float16* sA, _Float16* sW2 /* [2][HDIM*WP] */, float* sBias,
    const _Float16* __restrict__ wt, const float* __restrict__ b,
    int kTot, int tid, int lane, int wv)
{
    const int half  = lane >> 4;
    const int l15   = lane & 15;
    const int m0    = (wv >> 1) * 32;
    const int nBase = (wv & 1) * 128;
    const int Kdw   = kTot >> 1;
    const int nch   = kTot >> 6;

    sBias[tid] = b[tid];                 // previous layer ended with a barrier

    v8f zero = {};
    v8f acc[2][8];
#pragma unroll
    for (int mi = 0; mi < 2; ++mi)
#pragma unroll
        for (int nt = 0; nt < 8; ++nt) acc[mi][nt] = zero;

    if (wv == 0)
        tdm_issue_chunk(wt, (unsigned)(unsigned long long)(void*)sW2, Kdw);

    for (int c = 0; c < nch; ++c) {
        __syncthreads();                 // buf[(c+1)&1] consumers (chunk c-1) done
        if (wv == 0) {
            if (c + 1 < nch) {
                tdm_issue_chunk(wt + (size_t)(c + 1) * 64,
                                (unsigned)(unsigned long long)(void*)(sW2 + ((c + 1) & 1) * (HDIM * WP)),
                                Kdw);
                __builtin_amdgcn_s_wait_tensorcnt(1);   // chunk c landed
            } else {
                __builtin_amdgcn_s_wait_tensorcnt(0);
            }
        }
        __syncthreads();
        gemm_ksteps(sA, sW2 + (c & 1) * (HDIM * WP), acc, c * 64, m0, nBase, l15, half);
    }
    __syncthreads();                     // all reads of sA complete
    writeback(sA, sBias, acc, m0, nBase, l15, half);
    __syncthreads();
}

// ---------------------------------------------------------------------------
// Fallback layer: f32 weights from global, transpose+cvt in-kernel
// ---------------------------------------------------------------------------
__device__ __forceinline__ void run_layer_direct(
    _Float16* sA, _Float16* sW, float* sBias,
    const float* __restrict__ w, const float* __restrict__ b,
    int d_in, int kTot, int tid, int lane, int wv)
{
    const int half  = lane >> 4;
    const int l15   = lane & 15;
    const int m0    = (wv >> 1) * 32;
    const int nBase = (wv & 1) * 128;

    sBias[tid] = b[tid];

    v8f zero = {};
    v8f acc[2][8];
#pragma unroll
    for (int mi = 0; mi < 2; ++mi)
#pragma unroll
        for (int nt = 0; nt < 8; ++nt) acc[mi][nt] = zero;

    const int nchunks = kTot >> 6;
    for (int c = 0; c < nchunks; ++c) {
        __syncthreads();
        {
            _Float16* dst = sW + tid * WP;
            for (int kk = 0; kk < 64; ++kk) {
                int k = c * 64 + kk;
                float v = (k < d_in) ? w[(size_t)k * HDIM + tid] : 0.0f;
                dst[kk] = (_Float16)v;
            }
        }
        __syncthreads();
        gemm_ksteps(sA, sW, acc, c * 64, m0, nBase, l15, half);
    }
    __syncthreads();
    writeback(sA, sBias, acc, m0, nBase, l15, half);
    __syncthreads();
}

// ---------------------------------------------------------------------------
// Shared head (positional encoding) and tail (256 -> 3)
// ---------------------------------------------------------------------------
__device__ __forceinline__ void pos_encode(_Float16* sA, const float* coords,
                                           int row0, int tid)
{
    if (tid < MB) {
        const int r = tid;
        const float c0 = coords[(size_t)(row0 + r) * 2 + 0];
        const float c1 = coords[(size_t)(row0 + r) * 2 + 1];
        _Float16* pa = sA + r * AP;
        pa[0] = (_Float16)c0;
        pa[1] = (_Float16)c1;
        float f = 3.14159265358979323846f;
#pragma unroll
        for (int i = 0; i < 13; ++i) {
            const float a0 = c0 * f, a1 = c1 * f;
            pa[2 + 4 * i + 0] = (_Float16)__sinf(a0);
            pa[2 + 4 * i + 1] = (_Float16)__cosf(a0);
            pa[2 + 4 * i + 2] = (_Float16)__sinf(a1);
            pa[2 + 4 * i + 3] = (_Float16)__cosf(a1);
            f *= 2.0f;
        }
        for (int k = 54; k < 64; ++k) pa[k] = (_Float16)0.0f;
    }
    __syncthreads();
}

__device__ __forceinline__ void final_head(const _Float16* sA, float* sWl, float* sBl,
                                           const float* wl, const float* bl,
                                           float* out, int row0, int tid)
{
    sWl[tid]       = wl[tid];
    sWl[tid + 256] = wl[tid + 256];
    sWl[tid + 512] = wl[tid + 512];
    if (tid < 3) sBl[tid] = bl[tid];
    __syncthreads();
    if (tid < MB) {
        const int r = tid;
        float a0 = sBl[0], a1 = sBl[1], a2 = sBl[2];
        const _Float16* pa = sA + r * AP;
        for (int k = 0; k < HDIM; ++k) {
            const float x = (float)pa[k];
            a0 += x * sWl[k];
            a1 += x * sWl[256 + k];
            a2 += x * sWl[512 + k];
        }
        float* po = out + (size_t)(row0 + r) * 3;
        po[0] = a0; po[1] = a1; po[2] = a2;
    }
}

// ws layout (f16 elements): L0 tile @0 (256x64), then L1..L4 tiles (256x256 each)
#define WS_OFF0 0
#define WS_OFF1 (16384)
#define WS_OFF2 (WS_OFF1 + 4   * 65536)
#define WS_OFF3 (WS_OFF2 + 16  * 65536)
#define WS_OFF4 (WS_OFF3 + 64  * 65536)
#define WS_TOTAL_F16 (WS_OFF4 + 256 * 65536)

__global__ __launch_bounds__(256) void loe_fused_tdm(
    const float* __restrict__ coords, const _Float16* __restrict__ wt,
    const float* __restrict__ b0, const float* __restrict__ b1,
    const float* __restrict__ b2, const float* __restrict__ b3,
    const float* __restrict__ b4,
    const float* __restrict__ wl, const float* __restrict__ bl,
    float* __restrict__ out)
{
    __shared__ _Float16 sA[MB * AP];
    __shared__ _Float16 sW[2 * HDIM * WP];   // double-buffered TDM destination
    __shared__ float    sBias[HDIM];
    __shared__ float    sWl[3 * HDIM];
    __shared__ float    sBl[4];

    const int tid  = threadIdx.x;
    const int lane = tid & 31;
    const int wv   = tid >> 5;
    const int row0 = blockIdx.x * MB;

    pos_encode(sA, coords, row0, tid);

    run_layer_tdm(sA, sW, sBias, wt + WS_OFF0,                                   b0,  64, tid, lane, wv);
    run_layer_tdm(sA, sW, sBias, wt + WS_OFF1 + (size_t)(row0 >> 16) * 65536u,   b1, 256, tid, lane, wv);
    run_layer_tdm(sA, sW, sBias, wt + WS_OFF2 + (size_t)(row0 >> 14) * 65536u,   b2, 256, tid, lane, wv);
    run_layer_tdm(sA, sW, sBias, wt + WS_OFF3 + (size_t)(row0 >> 12) * 65536u,   b3, 256, tid, lane, wv);
    run_layer_tdm(sA, sW, sBias, wt + WS_OFF4 + (size_t)(row0 >> 10) * 65536u,   b4, 256, tid, lane, wv);

    final_head(sA, sWl, sBl, wl, bl, out, row0, tid);
}

__global__ __launch_bounds__(256) void loe_fused_direct(
    const float* __restrict__ coords,
    const float* __restrict__ w0, const float* __restrict__ b0,
    const float* __restrict__ w1, const float* __restrict__ b1,
    const float* __restrict__ w2, const float* __restrict__ b2,
    const float* __restrict__ w3, const float* __restrict__ b3,
    const float* __restrict__ w4, const float* __restrict__ b4,
    const float* __restrict__ wl, const float* __restrict__ bl,
    float* __restrict__ out)
{
    __shared__ _Float16 sA[MB * AP];
    __shared__ _Float16 sW[HDIM * WP];
    __shared__ float    sBias[HDIM];
    __shared__ float    sWl[3 * HDIM];
    __shared__ float    sBl[4];

    const int tid  = threadIdx.x;
    const int lane = tid & 31;
    const int wv   = tid >> 5;
    const int row0 = blockIdx.x * MB;

    pos_encode(sA, coords, row0, tid);

    run_layer_direct(sA, sW, sBias, w0,                                 b0,  54,  64, tid, lane, wv);
    run_layer_direct(sA, sW, sBias, w1 + (size_t)(row0 >> 16) * 65536u, b1, 256, 256, tid, lane, wv);
    run_layer_direct(sA, sW, sBias, w2 + (size_t)(row0 >> 14) * 65536u, b2, 256, 256, tid, lane, wv);
    run_layer_direct(sA, sW, sBias, w3 + (size_t)(row0 >> 12) * 65536u, b3, 256, 256, tid, lane, wv);
    run_layer_direct(sA, sW, sBias, w4 + (size_t)(row0 >> 10) * 65536u, b4, 256, 256, tid, lane, wv);

    final_head(sA, sWl, sBl, wl, bl, out, row0, tid);
}

extern "C" void kernel_launch(void* const* d_in, const int* in_sizes, int n_in,
                              void* d_out, int out_size, void* d_ws, size_t ws_size,
                              hipStream_t stream) {
    (void)in_sizes; (void)n_in; (void)out_size;
    const float* coords = (const float*)d_in[0];
    const float* w0 = (const float*)d_in[1];  const float* b0 = (const float*)d_in[2];
    const float* w1 = (const float*)d_in[3];  const float* b1 = (const float*)d_in[4];
    const float* w2 = (const float*)d_in[5];  const float* b2 = (const float*)d_in[6];
    const float* w3 = (const float*)d_in[7];  const float* b3 = (const float*)d_in[8];
    const float* w4 = (const float*)d_in[9];  const float* b4 = (const float*)d_in[10];
    const float* wl = (const float*)d_in[11]; const float* bl = (const float*)d_in[12];

    const int nblocks = 262144 / MB;   // 2048 workgroups of 256 threads (8 wave32)

    if (ws_size >= (size_t)WS_TOTAL_F16 * sizeof(_Float16)) {
        _Float16* wt = (_Float16*)d_ws;
        dim3 cb(32, 8);
        convert_transpose_kernel<<<dim3(2, 8, 1),   cb, 0, stream>>>(w0, wt + WS_OFF0,  54,  64);
        convert_transpose_kernel<<<dim3(8, 8, 4),   cb, 0, stream>>>(w1, wt + WS_OFF1, 256, 256);
        convert_transpose_kernel<<<dim3(8, 8, 16),  cb, 0, stream>>>(w2, wt + WS_OFF2, 256, 256);
        convert_transpose_kernel<<<dim3(8, 8, 64),  cb, 0, stream>>>(w3, wt + WS_OFF3, 256, 256);
        convert_transpose_kernel<<<dim3(8, 8, 256), cb, 0, stream>>>(w4, wt + WS_OFF4, 256, 256);
        loe_fused_tdm<<<nblocks, 256, 0, stream>>>(
            coords, wt, b0, b1, b2, b3, b4, wl, bl, (float*)d_out);
    } else {
        loe_fused_direct<<<nblocks, 256, 0, stream>>>(
            coords, w0, b0, w1, b1, w2, b2, w3, b3, w4, b4, wl, bl, (float*)d_out);
    }
}
